// Net_77618648973637
// MI455X (gfx1250) — matrix operand order repaired and checked by reading, not compiled, not run on hardware
//
#include <hip/hip_runtime.h>

// ---------------------------------------------------------------------------
// Types for WMMA (gfx1250, wave32): v16bf A/B fragments, v8f accumulator
// ---------------------------------------------------------------------------
typedef __attribute__((ext_vector_type(16))) __bf16        v16bf;
typedef __attribute__((ext_vector_type(8)))  float         v8f;
typedef __attribute__((ext_vector_type(4)))  unsigned int  u32x4;
typedef __attribute__((ext_vector_type(2)))  unsigned int  u32x2;

#define N_NODES  10000
#define N_EDGES  160000
#define IN_F     256
#define CH       512
#define N_LABELS 1440

__device__ __forceinline__ unsigned short f2bf(float f) {
    unsigned int u = __float_as_uint(f);
    unsigned int r = u + 0x7FFFu + ((u >> 16) & 1u);   // round-to-nearest-even
    return (unsigned short)(r >> 16);
}

// ---------------------------------------------------------------------------
// Graph preprocessing kernels (deterministic CSR build)
// ---------------------------------------------------------------------------
__global__ void count_degrees(const int* __restrict__ src, const int* __restrict__ dst,
                              int* __restrict__ deg_out, int* __restrict__ deg_in, int E) {
    int e = blockIdx.x * blockDim.x + threadIdx.x;
    if (e < E) {
        atomicAdd(&deg_out[src[e]], 1);
        atomicAdd(&deg_in[dst[e]], 1);
    }
}

__global__ __launch_bounds__(1024) void scan_kernel(const int* __restrict__ cnt,
                                                    int* __restrict__ starts,
                                                    int* __restrict__ cursor, int n) {
    __shared__ int s[1024];
    __shared__ int carry;
    int t = threadIdx.x;
    if (t == 0) carry = 0;
    __syncthreads();
    for (int base = 0; base < n; base += 1024) {
        int idx = base + t;
        int v = (idx < n) ? cnt[idx] : 0;
        s[t] = v;
        __syncthreads();
        for (int off = 1; off < 1024; off <<= 1) {   // Hillis-Steele inclusive scan
            int x = (t >= off) ? s[t - off] : 0;
            __syncthreads();
            s[t] += x;
            __syncthreads();
        }
        int incl = s[t];
        int excl = incl - v;
        if (idx < n) { starts[idx] = carry + excl; cursor[idx] = carry + excl; }
        __syncthreads();
        if (t == 1023) carry += incl;                // chunk total
        __syncthreads();
    }
    if (t == 0) starts[n] = carry;
}

__global__ void fill_kernel(const int* __restrict__ src, int* __restrict__ cursor,
                            int* __restrict__ edge_of, int E) {
    int e = blockIdx.x * blockDim.x + threadIdx.x;
    if (e < E) {
        int p = atomicAdd(&cursor[src[e]], 1);
        edge_of[p] = e;
    }
}

// sort each node's edge-id segment ascending -> deterministic float sum order
__global__ void sort_segments(const int* __restrict__ starts, int* __restrict__ edge_of, int n) {
    int v = blockIdx.x * blockDim.x + threadIdx.x;
    if (v >= n) return;
    int beg = starts[v], end = starts[v + 1];
    for (int i = beg + 1; i < end; i++) {
        int key = edge_of[i];
        int j = i - 1;
        while (j >= beg && edge_of[j] > key) { edge_of[j + 1] = edge_of[j]; j--; }
        edge_of[j + 1] = key;
    }
}

__global__ void edge_weights(const int* __restrict__ src, const int* __restrict__ dst,
                             const int* __restrict__ deg_out, const int* __restrict__ deg_in,
                             float* __restrict__ w, int E) {
    int e = blockIdx.x * blockDim.x + threadIdx.x;
    if (e < E) {
        float d0 = (float)max(deg_out[src[e]], 1);
        float d1 = (float)max(deg_in[dst[e]], 1);
        w[e] = rsqrtf(d0 * d1);
    }
}

__global__ void cvt_f32_bf16(const float* __restrict__ in, unsigned short* __restrict__ out, int n) {
    for (int i = blockIdx.x * blockDim.x + threadIdx.x; i < n; i += gridDim.x * blockDim.x)
        out[i] = f2bf(in[i]);
}

// ---------------------------------------------------------------------------
// Gather aggregation: agg[v] = sum_{e: src[e]==v} w[e] * m[dst[e]]
// one block per node, 128 threads x float4 cover 512 channels with b128 loads.
// m (20 MB) is L2-resident -> edge row reads hit the 192 MB L2.
// ---------------------------------------------------------------------------
__global__ __launch_bounds__(128) void aggregate(const float* __restrict__ m,
                                                 const int* __restrict__ starts,
                                                 const int* __restrict__ edge_of,
                                                 const int* __restrict__ dst,
                                                 const float* __restrict__ w,
                                                 float* __restrict__ agg) {
    int v = blockIdx.x;
    int t = threadIdx.x;
    int beg = starts[v], end = starts[v + 1];
    float ax = 0.0f, ay = 0.0f, az = 0.0f, aw = 0.0f;
    for (int i = beg; i < end; i++) {
        int e = edge_of[i];
        float we = w[e];
        const float4* mr = (const float4*)(m + (size_t)dst[e] * CH);
        if (i + 1 < end) {   // prefetch next edge's m row (global_prefetch_b8)
            int en = edge_of[i + 1];
            __builtin_prefetch(m + (size_t)dst[en] * CH + t * 4, 0, 0);
        }
        float4 mv = mr[t];
        ax += we * mv.x; ay += we * mv.y; az += we * mv.z; aw += we * mv.w;
    }
    float4 res; res.x = ax; res.y = ay; res.z = az; res.w = aw;
    ((float4*)(agg + (size_t)v * CH))[t] = res;
}

// ---------------------------------------------------------------------------
// WMMA bf16 GEMM: C[M,NCOLS] = A[M,K] @ B[K,NCOLS], bf16 in, f32 accumulate.
// Block: 256 threads = 8 waves; tile 256x64; K-step 32; double-buffered LDS.
// Wave w owns rows [w*32, w*32+32): 2 A fragments x 4 B fragments = 8 WMMA/step.
// A tile staged with CDNA5 async global->LDS (global_load_async_to_lds_b128,
// GVS mode), overlapped with WMMA on the other buffer; B tile (4KB) staged
// through VGPRs with the WMMA fragment swizzle (packed b32 LDS stores).
// NCOLS is compile-time -> all row*N address math strength-reduced.
// MODE 0: C = A@B                     -> f32 Cf
// MODE 1: C = relu(agg + A@B + bias)  -> bf16 Cb   (fused ARMA epilogue)
// MODE 2: C = A@B + bias              -> f32 Cf    (final classifier)
// ---------------------------------------------------------------------------
template <int MODE, int NCOLS>
__global__ __launch_bounds__(256) void gemm_wmma(const unsigned short* __restrict__ A,
                                                 const unsigned short* __restrict__ B,
                                                 int M, int K,
                                                 const float* __restrict__ bias,
                                                 const float* __restrict__ agg,
                                                 float* __restrict__ Cf,
                                                 unsigned short* __restrict__ Cb) {
    // A: 256 rows x 32 cols, padded row stride 40 (conflict-free frag reads)
    __shared__ __attribute__((aligned(16))) unsigned short sA[2][256 * 40];
    // B: fragment-major swizzle [frag f][lane][j], 4*32*16 bf16 per buffer
    __shared__ __attribute__((aligned(16))) unsigned short sB[2][4 * 32 * 16];

    const int t    = threadIdx.x;
    const int wave = t >> 5;
    const int lane = t & 31;
    const int ln   = lane & 15;
    const int hi   = (lane >> 4) & 1;
    const int row0 = blockIdx.x * 256;
    const int nb   = blockIdx.y * 64;

    v8f acc[2][4];
#pragma unroll
    for (int g = 0; g < 2; g++)
#pragma unroll
        for (int f = 0; f < 4; f++)
#pragma unroll
            for (int vv = 0; vv < 8; vv++) acc[g][f][vv] = 0.0f;

    // ---- A stage: 1024 x 16B quanta via async DMA; OOB rows clamped (their
    // results are discarded in the epilogue), so EXEC stays uniform.
    auto stageA = [&](int buf, int kt) {
#pragma unroll
        for (int i = t; i < 1024; i += 256) {
            int r = i >> 2, seg = (i & 3) * 8;
            int gr = row0 + r;
            if (gr >= M) gr = M - 1;
            unsigned int goff = ((unsigned int)gr * (unsigned int)K +
                                 (unsigned int)(kt + seg)) * 2u;
            unsigned int lds = (unsigned int)(size_t)&sA[buf][r * 40 + seg];
            asm volatile("global_load_async_to_lds_b128 %0, %1, %2"
                         :: "v"(lds), "v"(goff), "s"(A) : "memory");
        }
    };

    // ---- B stage: thread t reads 4 cols x 2 adjacent k rows, writes packed
    // (k,k+1) pairs as b32 into the fragment swizzle:
    //   lane L of frag f holds K=(j&7)+8*(L>=16)+16*(j>=8), N=f*16+(L&15).
    auto stageB = [&](int buf, int kt) {
        int k0 = (t >> 4) * 2;          // 0,2,..,30
        int n4 = (t & 15) * 4;          // 0,4,..,60
        u32x2 lo = {0u, 0u}, hi2 = {0u, 0u};
        if (nb + n4 < NCOLS) {
            lo  = *(const u32x2*)(B + (size_t)(kt + k0) * NCOLS + nb + n4);
            hi2 = *(const u32x2*)(B + (size_t)(kt + k0 + 1) * NCOLS + nb + n4);
        }
        const unsigned short* pl = (const unsigned short*)&lo;
        const unsigned short* ph = (const unsigned short*)&hi2;
        int f   = n4 >> 4;
        int j0  = (k0 & 7) | ((k0 >> 4) << 3);
        int lhs = (k0 & 8) ? 16 : 0;
        unsigned int* sb32 = (unsigned int*)&sB[buf][0];
#pragma unroll
        for (int q = 0; q < 4; q++) {
            int n = n4 + q;
            unsigned int val = (unsigned int)pl[q] | ((unsigned int)ph[q] << 16);
            sb32[(((f * 32) + ((n & 15) | lhs)) * 16 + j0) >> 1] = val;
        }
    };

    stageA(0, 0);
    stageB(0, 0);
    asm volatile("s_wait_asynccnt 0x0" ::: "memory");
    __syncthreads();

    const int nsteps = K >> 5;
    for (int s = 0; s < nsteps; s++) {
        const int cur = s & 1;
        const int nxt = cur ^ 1;
        if (s + 1 < nsteps) {           // overlap next-tile DMA with WMMA
            stageA(nxt, (s + 1) * 32);
            stageB(nxt, (s + 1) * 32);
        }

        // A fragments (ISA 16-bit A 16x32 layout):
        //  hi=0: elems 0..7 = K 0..7,  8..15 = K 16..23
        //  hi=1: elems 0..7 = K 8..15, 8..15 = K 24..31
        const unsigned short* a0 = &sA[cur][(wave * 32 + ln) * 40 + 8 * hi];
        const unsigned short* a1 = a0 + 16 * 40;
        v16bf av0, av1;
        ((u32x4*)&av0)[0] = *(const u32x4*)(a0);
        ((u32x4*)&av0)[1] = *(const u32x4*)(a0 + 16);
        ((u32x4*)&av1)[0] = *(const u32x4*)(a1);
        ((u32x4*)&av1)[1] = *(const u32x4*)(a1 + 16);
#pragma unroll
        for (int f = 0; f < 4; f++) {
            const unsigned short* bptr = &sB[cur][(f * 32 + lane) * 16];
            v16bf bv;
            ((u32x4*)&bv)[0] = *(const u32x4*)(bptr);
            ((u32x4*)&bv)[1] = *(const u32x4*)(bptr + 8);
            acc[0][f] = __builtin_amdgcn_wmma_f32_16x16x32_bf16(
                false, av0, false, bv, (short)0, acc[0][f], false, false);
            acc[1][f] = __builtin_amdgcn_wmma_f32_16x16x32_bf16(
                false, av1, false, bv, (short)0, acc[1][f], false, false);
        }

        if (s + 1 < nsteps)
            asm volatile("s_wait_asynccnt 0x0" ::: "memory");
        __syncthreads();
    }

    // ---- epilogue (C/D layout: VGPR v -> row = v + 8*hi, col = lane&15) ----
    // Fast path (all 256 rows valid, 39 of 40 blocks): incremental pointers,
    // no per-row compares. Slow path only for the final partial M-block.
    const int  colb = nb + ln;
    const bool full = (row0 + 256 <= M);
#pragma unroll
    for (int g = 0; g < 2; g++) {
        const int rowb = row0 + wave * 32 + g * 16 + 8 * hi;
        const size_t rbase = (size_t)rowb * NCOLS;
#pragma unroll
        for (int f = 0; f < 4; f++) {
            const int c = colb + f * 16;
            if (c >= NCOLS) continue;
            if (full) {
                if (MODE == 0) {
                    float* p = Cf + rbase + c;
#pragma unroll
                    for (int vv = 0; vv < 8; vv++) { *p = acc[g][f][vv]; p += NCOLS; }
                } else if (MODE == 1) {
                    const float* pa = agg + rbase + c;
                    unsigned short* pb = Cb + rbase + c;
                    const float bc = bias[c];
#pragma unroll
                    for (int vv = 0; vv < 8; vv++) {
                        float u = acc[g][f][vv] + *pa + bc;
                        *pb = f2bf(fmaxf(u, 0.0f));
                        pa += NCOLS; pb += NCOLS;
                    }
                } else {
                    float* p = Cf + rbase + c;
                    const float bc = bias[c];
#pragma unroll
                    for (int vv = 0; vv < 8; vv++) { *p = acc[g][f][vv] + bc; p += NCOLS; }
                }
            } else {
                const float bc = (MODE != 0) ? bias[c] : 0.0f;
#pragma unroll
                for (int vv = 0; vv < 8; vv++) {
                    int r = rowb + vv;
                    if (r >= M) continue;
                    float val = acc[g][f][vv];
                    if (MODE == 0) {
                        Cf[(size_t)r * NCOLS + c] = val;
                    } else if (MODE == 1) {
                        float u = val + agg[(size_t)r * NCOLS + c] + bc;
                        Cb[(size_t)r * NCOLS + c] = f2bf(fmaxf(u, 0.0f));
                    } else {
                        Cf[(size_t)r * NCOLS + c] = val + bc;
                    }
                }
            }
        }
    }
}

// ---------------------------------------------------------------------------
// Host-side orchestration
// ---------------------------------------------------------------------------
extern "C" void kernel_launch(void* const* d_in, const int* in_sizes, int n_in,
                              void* d_out, int out_size, void* d_ws, size_t ws_size,
                              hipStream_t stream) {
    const float* x   = (const float*)d_in[0];
    const int*   src = (const int*)d_in[1];
    const int*   dst = (const int*)d_in[2];
    const float* W1  = (const float*)d_in[3];
    const float* V1  = (const float*)d_in[4];
    const float* b1  = (const float*)d_in[5];
    const float* Wk  = (const float*)d_in[6];
    const float* Vk  = (const float*)d_in[7];
    const float* bk  = (const float*)d_in[8];
    const float* Wd  = (const float*)d_in[9];
    const float* bd  = (const float*)d_in[10];
    float* out = (float*)d_out;

    // ---- carve workspace (256B aligned slices) ----
    size_t off = 0;
    char* base = (char*)d_ws;
    auto alloc = [&](size_t bytes) -> void* {
        void* p = base + off;
        off += (bytes + 255) & ~(size_t)255;
        return p;
    };
    int*   deg_out_i = (int*)alloc((size_t)N_NODES * 4);
    int*   deg_in_i  = (int*)alloc((size_t)N_NODES * 4);
    int*   row_start = (int*)alloc((size_t)(N_NODES + 1) * 4);
    int*   cursor    = (int*)alloc((size_t)N_NODES * 4);
    int*   edge_of   = (int*)alloc((size_t)N_EDGES * 4);
    float* w         = (float*)alloc((size_t)N_EDGES * 4);
    unsigned short* x_bf = (unsigned short*)alloc((size_t)N_NODES * IN_F * 2);
    unsigned short* h0   = (unsigned short*)alloc((size_t)N_NODES * CH * 2);
    unsigned short* h1   = (unsigned short*)alloc((size_t)N_NODES * CH * 2);
    float* m   = (float*)alloc((size_t)N_NODES * CH * 4);
    float* agg = (float*)alloc((size_t)N_NODES * CH * 4);
    unsigned short* W1b = (unsigned short*)alloc((size_t)IN_F * CH * 2);
    unsigned short* V1b = (unsigned short*)alloc((size_t)IN_F * CH * 2);
    unsigned short* Wkb = (unsigned short*)alloc((size_t)6 * CH * CH * 2);
    unsigned short* Vkb = (unsigned short*)alloc((size_t)6 * CH * CH * 2);
    unsigned short* Wdb = (unsigned short*)alloc((size_t)CH * N_LABELS * 2);

    const int EB = (N_EDGES + 255) / 256;

    // ---- deterministic CSR build + edge weights ----
    hipMemsetAsync(deg_out_i, 0, (size_t)N_NODES * 4, stream);
    hipMemsetAsync(deg_in_i, 0, (size_t)N_NODES * 4, stream);
    count_degrees<<<EB, 256, 0, stream>>>(src, dst, deg_out_i, deg_in_i, N_EDGES);
    scan_kernel<<<1, 1024, 0, stream>>>(deg_out_i, row_start, cursor, N_NODES);
    fill_kernel<<<EB, 256, 0, stream>>>(src, cursor, edge_of, N_EDGES);
    sort_segments<<<(N_NODES + 255) / 256, 256, 0, stream>>>(row_start, edge_of, N_NODES);
    edge_weights<<<EB, 256, 0, stream>>>(src, dst, deg_out_i, deg_in_i, w, N_EDGES);

    // ---- f32 -> bf16 conversions (inputs + all weights, once per call) ----
    auto cvt = [&](const float* in, unsigned short* o, int n) {
        int blocks = (n + 255) / 256;
        if (blocks > 4096) blocks = 4096;
        cvt_f32_bf16<<<blocks, 256, 0, stream>>>(in, o, n);
    };
    cvt(x, x_bf, N_NODES * IN_F);
    cvt(W1, W1b, IN_F * CH);
    cvt(V1, V1b, IN_F * CH);
    cvt(Wk, Wkb, 6 * CH * CH);
    cvt(Vk, Vkb, 6 * CH * CH);
    cvt(Wd, Wdb, CH * N_LABELS);

    // ---- 7 ARMA layers ----
    const dim3 blk(256);
    const dim3 g512((N_NODES + 255) / 256, CH / 64);          // (40, 8)
    const unsigned short* hb = x_bf;
    int Kcur = IN_F;
    for (int l = 0; l < 7; l++) {
        const unsigned short* Wl = (l == 0) ? W1b : (Wkb + (size_t)(l - 1) * CH * CH);
        const unsigned short* Vl = (l == 0) ? V1b : (Vkb + (size_t)(l - 1) * CH * CH);
        const float*          bl = (l == 0) ? b1 : (bk + (size_t)(l - 1) * CH);
        unsigned short* hnext = (l & 1) ? h1 : h0;

        // m = h @ W
        gemm_wmma<0, CH><<<g512, blk, 0, stream>>>(hb, Wl, N_NODES, Kcur,
                                                   nullptr, nullptr, m, nullptr);
        // agg[v] = sum_{src[e]==v} w[e] * m[dst[e]]
        aggregate<<<N_NODES, 128, 0, stream>>>(m, row_start, edge_of, dst, w, agg);
        // h' = relu(agg + h @ V + b)  (fused epilogue, bf16 output)
        gemm_wmma<1, CH><<<g512, blk, 0, stream>>>(hb, Vl, N_NODES, Kcur,
                                                   bl, agg, nullptr, hnext);
        hb = hnext;
        Kcur = CH;
    }

    // ---- final classifier: out = h @ Wd + bd ----
    const dim3 gout((N_NODES + 255) / 256, (N_LABELS + 63) / 64); // (40, 23)
    gemm_wmma<2, N_LABELS><<<gout, blk, 0, stream>>>(hb, Wdb, N_NODES, CH,
                                                     bd, nullptr, out, nullptr);
}
